// ContinuousReasoningNavigator_35132832481464
// MI455X (gfx1250) — matrix operand order repaired and checked by reading, not compiled, not run on hardware
//
#include <hip/hip_runtime.h>
#include <math.h>

// ---------------------------------------------------------------- constants
constexpr int kB  = 32768;   // batch
constexpr int kH  = 1024;    // hidden
constexpr int kR  = 256;     // reasoning dim
constexpr int kM  = 500;     // memory slots
constexpr int kMp = 512;     // padded memory slots

// ---------------------------------------------------------------- types
typedef __attribute__((ext_vector_type(16))) __bf16          v16bf;
typedef __attribute__((ext_vector_type(8)))  float           v8f;
typedef __attribute__((ext_vector_type(8)))  unsigned short  ush8;
typedef __attribute__((ext_vector_type(4)))  unsigned short  ush4;
typedef __attribute__((ext_vector_type(4)))  float           f32x4;
typedef int b128_t __attribute__((vector_size(16)));         // matches builtin param

union ABfrag { ush8 h[2]; v16bf v; };

__device__ __forceinline__ unsigned short f2bf(float f) {
  unsigned int u = __float_as_uint(f);
  u += 0x7FFFu + ((u >> 16) & 1u);          // round-to-nearest-even
  return (unsigned short)(u >> 16);
}

// gfx1250 async global->LDS copy (ASYNCcnt path), with safe fallback
#if defined(__HIP_DEVICE_COMPILE__) && __has_builtin(__builtin_amdgcn_global_load_async_to_lds_b128)
#define HAVE_ASYNC_LDS 1
#endif

__device__ __forceinline__ void cp_b128_g2l(const unsigned short* g, unsigned short* l) {
#ifdef HAVE_ASYNC_LDS
  __builtin_amdgcn_global_load_async_to_lds_b128(
      (__attribute__((address_space(1))) b128_t*)const_cast<unsigned short*>(g),
      (__attribute__((address_space(3))) b128_t*)l, 0, 0);
#else
  *(ush8*)l = *(const ush8*)g;
#endif
}

__device__ __forceinline__ void wait_async_lds() {
#ifdef HAVE_ASYNC_LDS
#if __has_builtin(__builtin_amdgcn_s_wait_asynccnt)
  __builtin_amdgcn_s_wait_asynccnt(0);
#else
  asm volatile("s_wait_asynccnt 0" ::: "memory");
#endif
#endif
}

// ---------------------------------------------------------------- WMMA GEMM
// C[M,N] = act(A[M,K] @ Wt[N,K]^T + bias). A is f32 or bf16 (compile-time);
// Wt is bf16 K-major. Block tile 128x128, BK=32, 256 threads = 8 waves; wave w
// owns rows w*16..+15, 8 WMMA column tiles each.
#define BM 128
#define BN 128
#define BK 32
#define LDT 40   // ushort stride (pad 8): 80B rows -> all fragments 16B aligned

template <bool A_BF16>
__global__ __launch_bounds__(256) void gemm_wmma_bf16(
    const void* __restrict__ Ain,
    const unsigned short* __restrict__ Wt,       // [N,K] bf16
    const float* __restrict__ bias,
    float* __restrict__ Cf, unsigned short* __restrict__ Cb,
    int M, int N, int K, int do_relu)
{
  __shared__ unsigned short As[BM * LDT];
  __shared__ unsigned short Bs[BN * LDT];

  const int tid  = threadIdx.x;
  const int lane = tid & 31;
  const int wave = tid >> 5;
  const int m16  = lane & 15;
  const int hi   = lane >> 4;

  const int rowBase = blockIdx.y * BM;
  const int colBase = blockIdx.x * BN;

  v8f acc[8] = {};

  for (int k0 = 0; k0 < K; k0 += BK) {
    if constexpr (A_BF16) {
      // ---- A tile: pure contiguous bf16 copy, 2x B128 per thread (async path)
      const unsigned short* A = (const unsigned short*)Ain;
      #pragma unroll
      for (int i = 0; i < 2; ++i) {
        int q = i * 256 + tid;
        int r = q >> 2, ch = q & 3;                     // 4x 8-elem chunks per row
        cp_b128_g2l(&A[(size_t)(rowBase + r) * K + k0 + ch * 8],
                    &As[r * LDT + ch * 8]);
      }
    } else {
      // ---- A tile: f32 -> bf16 convert, vectorized float4 loads
      const float* A = (const float*)Ain;
      #pragma unroll
      for (int i = 0; i < 4; ++i) {
        int q = i * 256 + tid;
        int r = q >> 3, c4 = q & 7;                     // 8x 4-elem chunks per row
        f32x4 v = *(const f32x4*)&A[(size_t)(rowBase + r) * K + k0 + c4 * 4];
        ush4 p;
        p[0] = f2bf(v[0]); p[1] = f2bf(v[1]); p[2] = f2bf(v[2]); p[3] = f2bf(v[3]);
        *(ush4*)&As[r * LDT + c4 * 4] = p;
      }
      if (k0 + BK < K)
        __builtin_prefetch(&A[(size_t)(rowBase + (tid >> 1)) * K + k0 + BK], 0, 1);
    }
    // ---- W tile: contiguous bf16 K-major copy, 2x B128 per thread (async path)
    #pragma unroll
    for (int i = 0; i < 2; ++i) {
      int q = i * 256 + tid;
      int r = q >> 2, ch = q & 3;
      cp_b128_g2l(&Wt[(size_t)(colBase + r) * K + k0 + ch * 8],
                  &Bs[r * LDT + ch * 8]);
    }
    wait_async_lds();
    __syncthreads();

    // ---- A fragment: 16x32 bf16 layout (lanes<16: K 0..7,16..23; lanes>=16: +8)
    ABfrag a;
    const int arow = wave * 16 + m16;
    a.h[0] = *(const ush8*)&As[arow * LDT + hi * 8];
    a.h[1] = *(const ush8*)&As[arow * LDT + 16 + hi * 8];

    #pragma unroll
    for (int t = 0; t < 8; ++t) {
      // ---- B fragment: 32x16 (lane = column; lanes<16: K 0..7,16..23; lanes>=16: +8)
      ABfrag b;
      const int brow = t * 16 + m16;
      b.h[0] = *(const ush8*)&Bs[brow * LDT + hi * 8];
      b.h[1] = *(const ush8*)&Bs[brow * LDT + 16 + hi * 8];
      acc[t] = __builtin_amdgcn_wmma_f32_16x16x32_bf16(
          false, a.v, false, b.v, (short)0, acc[t], false, false);
    }
    __syncthreads();
  }

  // ---- epilogue: C/D layout VGPR v -> row v + hi*8, lane%16 -> column
  #pragma unroll
  for (int t = 0; t < 8; ++t) {
    const int ncol = colBase + t * 16 + m16;
    const float bv = bias ? bias[ncol] : 0.0f;
    #pragma unroll
    for (int v = 0; v < 8; ++v) {
      const int row = rowBase + wave * 16 + v + hi * 8;
      float val = acc[t][v] + bv;
      if (do_relu) val = fmaxf(val, 0.0f);
      if (Cf) Cf[(size_t)row * N + ncol] = val;
      if (Cb) Cb[(size_t)row * N + ncol] = f2bf(val);
    }
  }
}

// ---------------------------------------------------------------- small kernels
// transpose+convert weight W[K,N] f32 -> Wt[N,K] bf16 (tiny, one block per row n)
__global__ __launch_bounds__(256) void wtrans_bf16_kernel(
    const float* __restrict__ W, unsigned short* __restrict__ Wt, int K, int N)
{
  const int n = blockIdx.x;
  for (int k = threadIdx.x; k < K; k += 256)
    Wt[(size_t)n * K + k] = f2bf(W[(size_t)k * N + n]);
}

// normalized memory bank -> nbTt[Np=512][K=256] bf16 (zero padded rows)
__global__ __launch_bounds__(256) void bank_norm_kernel(
    const float* __restrict__ bank, unsigned short* __restrict__ nbTt)
{
  __shared__ float sd[256];
  const int n = blockIdx.x;        // 0..511
  const int t = threadIdx.x;       // 0..255 == R
  float x = (n < kM) ? bank[n * kR + t] : 0.0f;
  sd[t] = x * x; __syncthreads();
  for (int s = 128; s > 0; s >>= 1) { if (t < s) sd[t] += sd[t + s]; __syncthreads(); }
  float inv = 1.0f / fmaxf(sqrtf(sd[0]), 1e-12f);
  nbTt[(size_t)n * kR + t] = f2bf((n < kM) ? x * inv : 0.0f);
}

__global__ __launch_bounds__(256) void rownorm_bf16_kernel(
    const float* __restrict__ src, unsigned short* __restrict__ dst)
{
  __shared__ float sd[256];
  const size_t b = blockIdx.x;
  const int t = threadIdx.x;
  float x = src[b * kR + t];
  sd[t] = x * x; __syncthreads();
  for (int s = 128; s > 0; s >>= 1) { if (t < s) sd[t] += sd[t + s]; __syncthreads(); }
  float inv = 1.0f / fmaxf(sqrtf(sd[0]), 1e-12f);
  dst[b * kR + t] = f2bf(x * inv);
}

__global__ __launch_bounds__(256) void topk3_kernel(
    const float* __restrict__ sims, const float* __restrict__ vals,
    int* __restrict__ idx)
{
  const int b = blockIdx.x * blockDim.x + threadIdx.x;
  if (b >= kB) return;
  const float* row = sims + (size_t)b * kMp;
  float w0 = -3.4e38f, w1 = -3.4e38f, w2 = -3.4e38f;
  int i0 = 0, i1 = 0, i2 = 0;
  for (int m = 0; m < kM; ++m) {
    float s = row[m] * (vals[m] + 1e-8f);
    if (s > w0)      { w2 = w1; i2 = i1; w1 = w0; i1 = i0; w0 = s; i0 = m; }
    else if (s > w1) { w2 = w1; i2 = i1; w1 = s;  i1 = m; }
    else if (s > w2) { w2 = s;  i2 = m; }
  }
  idx[b * 4 + 0] = i0; idx[b * 4 + 1] = i1; idx[b * 4 + 2] = i2;
}

__global__ __launch_bounds__(256) void fuse_kernel(
    const float* __restrict__ rs0, const float* __restrict__ bank,
    const int* __restrict__ idx, float* __restrict__ rs_f,
    unsigned short* __restrict__ rs_b)
{
  const size_t b = blockIdx.x;
  const int t = threadIdx.x;
  const int i0 = idx[b * 4 + 0], i1 = idx[b * 4 + 1], i2 = idx[b * 4 + 2];
  float ret = (bank[(size_t)i0 * kR + t] + bank[(size_t)i1 * kR + t] +
               bank[(size_t)i2 * kR + t]) * (1.0f / 3.0f);
  float r = 0.5f * rs0[b * kR + t] + 0.5f * ret;
  rs_f[b * kR + t] = r;
  rs_b[b * kR + t] = f2bf(r);
}

__global__ __launch_bounds__(256) void heads_kernel(
    const float* __restrict__ rs, const float* __restrict__ u,
    const float* __restrict__ Wc, const float* __restrict__ bc,
    const float* __restrict__ Wsp, const float* __restrict__ bsp,
    const float* __restrict__ Wv, const float* __restrict__ bv,
    float* __restrict__ out_action, float* __restrict__ out_logp,
    float* __restrict__ out_value, float* __restrict__ out_entropy,
    float* __restrict__ step)
{
  const int b = blockIdx.x * 8 + (threadIdx.x >> 5);   // one wave per row
  const int lane = threadIdx.x & 31;
  const float* r = rs + (size_t)b * kR;
  float c0 = 0.f, c1 = 0.f, sv = 0.f, vv = 0.f;
  for (int k = lane; k < kR; k += 32) {
    float x = r[k];
    c0 += x * Wc[2 * k]; c1 += x * Wc[2 * k + 1];
    sv += x * Wsp[k];    vv += x * Wv[k];
  }
  for (int off = 16; off; off >>= 1) {
    c0 += __shfl_xor(c0, off);
    c1 += __shfl_xor(c1, off);
    sv += __shfl_xor(sv, off);
    vv += __shfl_xor(vv, off);
  }
  if (lane == 0) {
    float l0 = c0 + bc[0], l1 = c1 + bc[1];
    float mx = fmaxf(l0, l1);
    float z  = logf(expf(l0 - mx) + expf(l1 - mx));
    float lp0 = l0 - mx - z, lp1 = l1 - mx - z;
    float p0 = expf(lp0), p1 = expf(lp1);
    int act = (u[b] >= p0) ? 1 : 0;
    out_action[b]  = (float)act;
    out_logp[b]    = act ? lp1 : lp0;
    out_entropy[b] = -(p0 * lp0 + p1 * lp1);
    out_value[b]   = vv + bv[0];
    step[b] = 2.0f / (1.0f + expf(-(sv + bsp[0])));
  }
}

// next_pos = rs + step * normalize(dir); in-place safe on dir (reg-held)
__global__ __launch_bounds__(256) void nextpos_kernel(
    const float* __restrict__ rs, float* __restrict__ dir_np,
    const float* __restrict__ step, unsigned short* __restrict__ np_b)
{
  __shared__ float sd[256];
  const size_t b = blockIdx.x;
  const int t = threadIdx.x;
  float d = dir_np[b * kR + t];
  sd[t] = d * d; __syncthreads();
  for (int s = 128; s > 0; s >>= 1) { if (t < s) sd[t] += sd[t + s]; __syncthreads(); }
  float inv = 1.0f / fmaxf(sqrtf(sd[0]), 1e-12f);
  float np = rs[b * kR + t] + step[b] * d * inv;
  dir_np[b * kR + t] = np;           // overwrite dir0 with next_pos (f32 output)
  np_b[b * kR + t]   = f2bf(np);
}

__global__ __launch_bounds__(256) void copy_bank_kernel(
    const float* __restrict__ bank, float* __restrict__ ob)
{
  ob[(size_t)blockIdx.x * kR + threadIdx.x] = bank[(size_t)blockIdx.x * kR + threadIdx.x];
}

__global__ __launch_bounds__(256) void copy_vals_kernel(
    const float* __restrict__ vals, float* __restrict__ ov)
{
  const int i = blockIdx.x * 256 + threadIdx.x;
  if (i < kM) ov[i] = vals[i];
}

__global__ __launch_bounds__(256) void colmean1_kernel(
    const float* __restrict__ np, float* __restrict__ part)
{
  const int t = threadIdx.x;
  const int r0 = blockIdx.x * 1024;
  float s = 0.f;
  for (int r = r0; r < r0 + 1024; ++r) s += np[(size_t)r * kR + t];
  part[blockIdx.x * kR + t] = s;
}

__global__ __launch_bounds__(256) void colmean2_kernel(
    const float* __restrict__ part, float* __restrict__ ob)
{
  const int t = threadIdx.x;
  float s = 0.f;
  for (int j = 0; j < 32; ++j) s += part[j * kR + t];
  ob[t] = s * (1.0f / (float)kB);
}

__global__ __launch_bounds__(256) void vmean1_kernel(
    const float* __restrict__ val, float* __restrict__ part)
{
  __shared__ float sd[256];
  const int t = threadIdx.x;
  sd[t] = val[blockIdx.x * 256 + t]; __syncthreads();
  for (int s = 128; s > 0; s >>= 1) { if (t < s) sd[t] += sd[t + s]; __syncthreads(); }
  if (t == 0) part[blockIdx.x] = sd[0];
}

__global__ __launch_bounds__(128) void vmean2_kernel(
    const float* __restrict__ part, float* __restrict__ ov)
{
  __shared__ float sd[128];
  const int t = threadIdx.x;
  sd[t] = part[t]; __syncthreads();
  for (int s = 64; s > 0; s >>= 1) { if (t < s) sd[t] += sd[t + s]; __syncthreads(); }
  if (t == 0) ov[0] = sd[0] * (1.0f / (float)kB);
}

// ---------------------------------------------------------------- launcher
extern "C" void kernel_launch(void* const* d_in, const int* in_sizes, int n_in,
                              void* d_out, int out_size, void* d_ws, size_t ws_size,
                              hipStream_t stream)
{
  (void)in_sizes; (void)n_in; (void)out_size; (void)ws_size;

  const float* state = (const float*)d_in[0];
  const float* bank  = (const float*)d_in[1];
  const float* vals  = (const float*)d_in[2];
  const float* u     = (const float*)d_in[3];
  const float* W1 = (const float*)d_in[4];  const float* b1 = (const float*)d_in[5];
  const float* W2 = (const float*)d_in[6];  const float* b2 = (const float*)d_in[7];
  const float* W3 = (const float*)d_in[8];  const float* b3 = (const float*)d_in[9];
  const float* W4 = (const float*)d_in[10]; const float* b4 = (const float*)d_in[11];
  const float* Wc = (const float*)d_in[12]; const float* bc = (const float*)d_in[13];
  const float* Wd = (const float*)d_in[14]; const float* bd = (const float*)d_in[15];
  const float* Wsp = (const float*)d_in[16]; const float* bsp = (const float*)d_in[17];
  const float* Wv = (const float*)d_in[18]; const float* bv = (const float*)d_in[19];

  // output layout (flat, return order)
  float* out       = (float*)d_out;
  float* o_latent  = out;
  float* o_next    = o_latent + (size_t)kB * kH;
  float* o_action  = o_next + (size_t)kB * kR;
  float* o_logp    = o_action + kB;
  float* o_value   = o_logp + kB;
  float* o_entropy = o_value + kB;
  float* o_bank    = o_entropy + kB;
  float* o_vals    = o_bank + (size_t)kM * kR;

  // workspace layout (16B-aligned chunks)
  char* wsc = (char*)d_ws;
  float* rsF           = (float*)wsc;                 wsc += (size_t)kB * kR * 4;  // rs (f32)
  unsigned short* Xbf  = (unsigned short*)wsc;        wsc += (size_t)kB * kR * 2;  // bf16 A ping
  unsigned short* Ybf  = (unsigned short*)wsc;        wsc += (size_t)kB * kR * 2;  // bf16 A pong
  unsigned short* Wt1  = (unsigned short*)wsc;        wsc += (size_t)kR * kH * 2;  // [256,1024]
  unsigned short* Wt2  = (unsigned short*)wsc;        wsc += (size_t)kR * kR * 2;
  unsigned short* Wtd  = (unsigned short*)wsc;        wsc += (size_t)kR * kR * 2;
  unsigned short* Wt3  = (unsigned short*)wsc;        wsc += (size_t)kR * kR * 2;
  unsigned short* Wt4  = (unsigned short*)wsc;        wsc += (size_t)kH * kR * 2;  // [1024,256]
  unsigned short* nbTt = (unsigned short*)wsc;        wsc += (size_t)kMp * kR * 2; // [512,256]
  int*   idx  = (int*)wsc;                            wsc += (size_t)kB * 4 * 4;
  float* step = (float*)wsc;                          wsc += (size_t)kB * 4;
  float* part = (float*)wsc;                          // 128 + 32*256 floats

  float* sims = o_latent;      // [B,512] sims scratch (dead before latent GEMM)
  float* scr  = o_next;        // rs0 then dir0 then final next_pos (sequential reuse)

  dim3 blk(256);

  // 0. weight prep: transpose+convert to bf16 K-major (tiny)
  wtrans_bf16_kernel<<<kR, blk, 0, stream>>>(W1, Wt1, kH, kR);
  wtrans_bf16_kernel<<<kR, blk, 0, stream>>>(W2, Wt2, kR, kR);
  wtrans_bf16_kernel<<<kR, blk, 0, stream>>>(Wd, Wtd, kR, kR);
  wtrans_bf16_kernel<<<kR, blk, 0, stream>>>(W3, Wt3, kR, kR);
  wtrans_bf16_kernel<<<kH, blk, 0, stream>>>(W4, Wt4, kR, kH);
  bank_norm_kernel<<<kMp, blk, 0, stream>>>(bank, nbTt);

  // 1. tmp1 = relu(state @ W1 + b1) -> bf16       [B,1024]x[1024,256]
  gemm_wmma_bf16<false><<<dim3(kR / BN, kB / BM), blk, 0, stream>>>(
      state, Wt1, b1, nullptr, Xbf, kB, kR, kH, 1);
  // 2. rs0 = tmp1 @ W2 + b2 -> f32 (next_pos region scratch)
  gemm_wmma_bf16<true><<<dim3(kR / BN, kB / BM), blk, 0, stream>>>(
      Xbf, Wt2, b2, scr, nullptr, kB, kR, kR, 0);
  // 3. ns = normalize(rs0) -> bf16
  rownorm_bf16_kernel<<<kB, blk, 0, stream>>>(scr, Xbf);
  // 4. sims = ns @ nbT -> f32 (latent region scratch)  [B,256]x[256,512]
  gemm_wmma_bf16<true><<<dim3(kMp / BN, kB / BM), blk, 0, stream>>>(
      Xbf, nbTt, nullptr, sims, nullptr, kB, kMp, kR, 0);
  // 5. value-weighted top-3
  topk3_kernel<<<kB / 256, blk, 0, stream>>>(sims, vals, idx);
  // 6. rs = 0.5*rs0 + 0.5*gather_mean(bank) -> f32 + bf16
  fuse_kernel<<<kB, blk, 0, stream>>>(scr, bank, idx, rsF, Xbf);
  // 7. heads
  heads_kernel<<<kB / 8, blk, 0, stream>>>(rsF, u, Wc, bc, Wsp, bsp, Wv, bv,
                                           o_action, o_logp, o_value, o_entropy, step);
  // 8. dir0 = rs @ Wd + bd -> f32 (next_pos region, rs0 dead)
  gemm_wmma_bf16<true><<<dim3(kR / BN, kB / BM), blk, 0, stream>>>(
      Xbf, Wtd, bd, scr, nullptr, kB, kR, kR, 0);
  // 9. next_pos = rs + step*normalize(dir0) -> f32 in place + bf16
  nextpos_kernel<<<kB, blk, 0, stream>>>(rsF, scr, step, Xbf);
  // 10. tmp2 = relu(next_pos @ W3 + b3) -> bf16
  gemm_wmma_bf16<true><<<dim3(kR / BN, kB / BM), blk, 0, stream>>>(
      Xbf, Wt3, b3, nullptr, Ybf, kB, kR, kR, 1);
  // 11. latent = tmp2 @ W4 + b4 -> f32 (overwrites sims scratch)
  gemm_wmma_bf16<true><<<dim3(kH / BN, kB / BM), blk, 0, stream>>>(
      Ybf, Wt4, b4, o_latent, nullptr, kB, kH, kR, 0);
  // 12. circular memory write
  copy_bank_kernel<<<kM, blk, 0, stream>>>(bank, o_bank);
  copy_vals_kernel<<<2, blk, 0, stream>>>(vals, o_vals);
  colmean1_kernel<<<32, blk, 0, stream>>>(o_next, part + 128);
  colmean2_kernel<<<1, blk, 0, stream>>>(part + 128, o_bank);
  vmean1_kernel<<<128, blk, 0, stream>>>(o_value, part);
  vmean2_kernel<<<1, 128, 0, stream>>>(part, o_vals);
}
